// hw_layer_86612310491885
// MI455X (gfx1250) — compile-verified, matrix operand order checked by reading, not run on hardware
//
#include <hip/hip_runtime.h>
#include <hip/hip_bf16.h>

typedef __attribute__((ext_vector_type(2)))  float    v2f;
typedef __attribute__((ext_vector_type(8)))  float    v8f;
typedef __attribute__((ext_vector_type(16))) _Float16 v16h;

#define CH 64   // channels
#define KB 64   // codebook bins per channel

// One wave handles 16 time-rows of one channel (16 x 64 outputs).
// D[m][n] = x[m] - evaluate[c][n] is produced by WMMA:
//   A[m][k] = 0.25 (all slots)  -> layout-agnostic
//   B[k][n] = -e[c][n] (all slots of column n = lane&15)
//   C[m][n] = x[m]
__global__ __launch_bounds__(256) void vq_softmax_kernel(
    const float* __restrict__ x,
    const float* __restrict__ evaluate,
    const float* __restrict__ focus,
    float* __restrict__ out,
    int nrows)  // nrows = B*T
{
    const int lane = threadIdx.x & 31;
    const int half = lane >> 4;        // 0: rows r, 1: rows r+8
    const int col  = lane & 15;        // N within 16-wide tile
    const int w    = (blockIdx.x * blockDim.x + threadIdx.x) >> 5;
    const int c    = w & (CH - 1);
    const int bt0  = (w >> 6) << 4;    // first of 16 time-rows for this wave
    if (bt0 >= nrows) return;          // wave-uniform: EXEC stays all-ones

    // ---- B tiles: -evaluate[c][j*16+col], replicated in both K slots ----
    v2f bm[4];
#pragma unroll
    for (int j = 0; j < 4; ++j) {
        float e = evaluate[c * KB + j * 16 + col];
        bm[j].x = -e; bm[j].y = -e;
    }

    // ---- C: x[m] broadcast over n (C/D layout: vgpr r -> row r + 8*half) ----
    const int mbase = bt0 + 8 * half;
    v8f cx;
#pragma unroll
    for (int r = 0; r < 8; ++r)
        cx[r] = x[(size_t)(mbase + r) * CH + c];

    // ---- 4 WMMAs: d[j][r] = x[m] - e[c][16j + col] ----
    v8f d[4];
#if __has_builtin(__builtin_amdgcn_wmma_f32_16x16x4_f32)
    v2f am; am.x = 0.25f; am.y = 0.25f;   // A[m][k] = 1/4 for all 4 k-slots
#pragma unroll
    for (int j = 0; j < 4; ++j)
        d[j] = __builtin_amdgcn_wmma_f32_16x16x4_f32(
            false, am, false, bm[j], (short)0, cx, false, false);
#else
    // Fallback: confirmed f16 WMMA. A = 1/32 in all 32 k-slots, B = -e (f16).
    v16h ah;
#pragma unroll
    for (int i = 0; i < 16; ++i) ah[i] = (_Float16)0.03125f;
#pragma unroll
    for (int j = 0; j < 4; ++j) {
        v16h bh;
        float e = evaluate[c * KB + j * 16 + col];
#pragma unroll
        for (int i = 0; i < 16; ++i) bh[i] = (_Float16)(-e);
        d[j] = __builtin_amdgcn_wmma_f32_16x16x32_f16(
            false, ah, false, bh, (short)0, cx, false, false);
    }
#endif

    // ---- per-row argmin + softmax; both 16-lane halves work independently ----
#pragma unroll
    for (int r = 0; r < 8; ++r) {
        float a0 = fabsf(d[0][r]);
        float a1 = fabsf(d[1][r]);
        float a2 = fabsf(d[2][r]);
        float a3 = fabsf(d[3][r]);

        // in-lane argmin over the 4 tiles (ties -> lower k, matching argmin)
        float v = a0; int k = col;
        bool t;
        t = (a1 < v); v = t ? a1 : v; k = t ? (16 + col) : k;
        t = (a2 < v); v = t ? a2 : v; k = t ? (32 + col) : k;
        t = (a3 < v); v = t ? a3 : v; k = t ? (48 + col) : k;

        // 16-lane butterfly argmin (xor masks < 16 never cross the halves)
#pragma unroll
        for (int s = 1; s < 16; s <<= 1) {
            float ov = __shfl_xor(v, s, 32);
            int   ok = __shfl_xor(k, s, 32);
            bool take = (ov < v) || ((ov == v) && (ok < k));
            v = take ? ov : v;
            k = take ? ok : k;
        }

        const float f = focus[c * KB + k];

        // stable softmax: max logit = -v*f  ->  exp((v - a)*f)
        float e0 = __expf((v - a0) * f);
        float e1 = __expf((v - a1) * f);
        float e2 = __expf((v - a2) * f);
        float e3 = __expf((v - a3) * f);

        float sum = (e0 + e1) + (e2 + e3);
#pragma unroll
        for (int s = 1; s < 16; s <<= 1)
            sum += __shfl_xor(sum, s, 32);
        const float inv = 1.0f / sum;

        // streaming (non-temporal) stores: written once, never re-read
        float* po = out + (size_t)(mbase + r) * (CH * KB) + (size_t)c * KB + col;
        __builtin_nontemporal_store(e0 * inv, po);
        __builtin_nontemporal_store(e1 * inv, po + 16);
        __builtin_nontemporal_store(e2 * inv, po + 32);
        __builtin_nontemporal_store(e3 * inv, po + 48);
    }
}

extern "C" void kernel_launch(void* const* d_in, const int* in_sizes, int n_in,
                              void* d_out, int out_size, void* d_ws, size_t ws_size,
                              hipStream_t stream) {
    const float* x        = (const float*)d_in[0];   // [B,T,C]
    const float* evaluate = (const float*)d_in[1];   // [C,K]
    const float* focus    = (const float*)d_in[2];   // [C,K]
    float*       out      = (float*)d_out;           // [B,T,C*K]

    const int nrows  = in_sizes[0] / CH;             // B*T = 32768
    const int ntiles = (nrows + 15) / 16;            // 16 rows per wave
    const int waves  = ntiles * CH;                  // one wave per (tile, channel)
    const int blocks = (waves + 7) / 8;              // 8 waves per 256-thread block

    vq_softmax_kernel<<<blocks, 256, 0, stream>>>(x, evaluate, focus, out, nrows);
}
